// MultiHeadAttention_11733850652966
// MI455X (gfx1250) — matrix operand order repaired and checked by reading, not compiled, not run on hardware
//
#include <hip/hip_runtime.h>

typedef _Float16 f16;
typedef __attribute__((ext_vector_type(16))) _Float16 v16h;
typedef __attribute__((ext_vector_type(8)))  _Float16 v8h;
typedef __attribute__((ext_vector_type(8)))  float    v8f;

#define DEV static __device__ __forceinline__

constexpr int Bc  = 2;
constexpr int Sc  = 2048;
constexpr int Dc  = 1024;
constexpr int Hc  = 16;
constexpr int DHc = 64;
constexpr int Mc  = Bc * Sc;   // 4096 rows for all GEMMs

// ---------------------------------------------------------------------------
// WMMA fragment helpers (wave32)
// A 16x32 (MxK): lanes 0-15 -> rows; halves [0:8]=K(k0..k0+7), [8:16]=K(k0+16..k0+23);
//                lanes 16-31 use k0+8. (ISA 7.12.2)
// B 32x16 (KxN): lane&15 -> column; same K order as A (k0..7,k16..23; lanes>=16: +8).
// ---------------------------------------------------------------------------
DEV v16h join16(v8h lo, v8h hi) {
  return __builtin_shufflevector(lo, hi, 0,1,2,3,4,5,6,7,8,9,10,11,12,13,14,15);
}

DEV v8f wmma32(v16h a, v16h b, v8f c) {
  return __builtin_amdgcn_wmma_f32_16x16x32_f16(false, a, false, b, (short)0, c,
                                                false, false);
}

// A fragment from global, row-major [*, lda]
DEV v16h load_afrag(const f16* __restrict__ A, int m0, int k0, int lda) {
  const int lane = threadIdx.x & 31;
  const int m = m0 + (lane & 15);
  const int k = k0 + ((lane >> 4) << 3);     // +8 for lanes 16..31
  const f16* p = A + (size_t)m * lda + k;
  return join16(*(const v8h*)p, *(const v8h*)(p + 16));
}

// B fragment from an LDS tile row n: halfs k0+8h..+7 and k0+16+8h..+7 (A order)
DEV v16h lds_bfrag(const f16* row_base) {
  const f16* p = row_base - (((threadIdx.x & 31) >> 4) << 3);
  return join16(*(const v8h*)p, *(const v8h*)(p + 16));
}

// ---------------------------------------------------------------------------
// Async global->LDS staging (CDNA5), ASYNCcnt-tracked
// ---------------------------------------------------------------------------
DEV void async_load_b128(const f16* lds_dst, const f16* gsrc) {
  unsigned int ldsoff = (unsigned int)(size_t)lds_dst;   // LDS offset = addr[31:0]
  unsigned long long ga = (unsigned long long)(size_t)gsrc;
  asm volatile("global_load_async_to_lds_b128 %0, %1, off"
               :: "v"(ldsoff), "v"(ga) : "memory");
}
DEV void wait_async0() {
  asm volatile("s_wait_asynccnt 0x0" ::: "memory");
}

// ---------------------------------------------------------------------------
// fp32 -> f16 conversion (grid-strided)
// ---------------------------------------------------------------------------
__global__ void cvt_f32_f16_kernel(const float* __restrict__ in,
                                   f16* __restrict__ out, int n) {
  for (int i = blockIdx.x * blockDim.x + threadIdx.x; i < n;
       i += gridDim.x * blockDim.x)
    out[i] = (f16)in[i];
}

// ---------------------------------------------------------------------------
// GEMM: C[M=4096, N=1024] = A[M,K=1024] * W[N,K]^T + bias
// 256 thr = 8 waves; block tile 256(M) x 64(N); wave tile 32M x 64N (8 accums).
// B tile (64n x 32k) staged in LDS via double-buffered async copies; A-frags
// straight from global (per-wave rows, L2-resident).
// mode 0: f16 head-major Q/K [B,H,S,DH]; mode 1: f16 Vt [B,H,DH,S];
// mode 2: f32 row-major out.
// ---------------------------------------------------------------------------
__global__ __launch_bounds__(256)
void gemm_wmma_kernel(const f16* __restrict__ A, const f16* __restrict__ W,
                      const float* __restrict__ bias, void* __restrict__ outp,
                      int mode) {
  __shared__ __align__(16) f16 Bt[2][64][40];   // 32 halfs + pad

  const int tid   = threadIdx.x;
  const int lane  = tid & 31;
  const int wave  = tid >> 5;
  const int hi    = lane >> 4;
  const int mTile = blockIdx.x * 256 + wave * 32;
  const int nTile = blockIdx.y * 64;

  // cooperative B loader: thread t -> row n=t>>2 (64 rows), chunk (t&3)*8 halfs
  const int ln = tid >> 2, lc = (tid & 3) * 8;
  const f16* gW = W + (size_t)(nTile + ln) * Dc + lc;

  async_load_b128(&Bt[0][ln][lc], gW);          // kb = 0

  v8f zf = {};
  v8f acc[2][4] = {{zf, zf, zf, zf}, {zf, zf, zf, zf}};

  for (int i = 0; i < Dc / 32; ++i) {
    const int kb = i * 32;
    wait_async0();
    __syncthreads();
    if (kb + 32 < Dc) async_load_b128(&Bt[(i + 1) & 1][ln][lc], gW + kb + 32);

    v16h a0 = load_afrag(A, mTile,      kb, Dc);
    v16h a1 = load_afrag(A, mTile + 16, kb, Dc);
#pragma unroll
    for (int c = 0; c < 4; ++c) {
      v16h b = lds_bfrag(&Bt[i & 1][c * 16 + (lane & 15)][hi * 16]);
      acc[0][c] = wmma32(a0, b, acc[0][c]);
      acc[1][c] = wmma32(a1, b, acc[1][c]);
    }
    __syncthreads();
  }

#pragma unroll
  for (int sub = 0; sub < 2; ++sub) {
    const int mBase = mTile + sub * 16 + hi * 8;
#pragma unroll
    for (int c = 0; c < 4; ++c) {
      const int n  = nTile + c * 16 + (lane & 15);
      const float bv = bias[n];
      if (mode == 0) {
        f16* out = (f16*)outp;
        const int h = n >> 6, d = n & 63;
#pragma unroll
        for (int r = 0; r < 8; ++r) {
          const int m = mBase + r;
          const int b = m >> 11, s = m & (Sc - 1);
          out[(((size_t)(b * Hc + h) * Sc + s) << 6) + d] =
              (f16)(acc[sub][c][r] + bv);
        }
      } else if (mode == 1) {
        f16* out = (f16*)outp;
        const int h = n >> 6, d = n & 63;
        const int b = mBase >> 11, s = mBase & (Sc - 1);
        v8h pk;
#pragma unroll
        for (int r = 0; r < 8; ++r) pk[r] = (f16)(acc[sub][c][r] + bv);
        *(v8h*)(out + ((size_t)(b * Hc + h) * DHc + d) * Sc + s) = pk;
      } else {
        float* out = (float*)outp;
#pragma unroll
        for (int r = 0; r < 8; ++r) {
          const int m = mBase + r;
          out[(size_t)m * Dc + n] = acc[sub][c][r] + bv;
        }
      }
    }
  }
}

// ---------------------------------------------------------------------------
// Flash attention: grid (B*H, S/128), 256 thr = 8 waves, wave owns 16 q rows.
// Per 32-key block: K tile (32x64) and V tile (64x32, transposed V) staged in
// LDS via double-buffered async copies and shared by all 8 waves.
//   scores: 4 wmma; online softmax via shfl_xor row reductions;
//   P via per-wave LDS tile (D-layout -> A-layout); PV: 4 wmma.
// ---------------------------------------------------------------------------
__global__ __launch_bounds__(256)
void attn_kernel(const f16* __restrict__ Qh, const f16* __restrict__ Kh,
                 const f16* __restrict__ Vt, const int* __restrict__ mask,
                 f16* __restrict__ outc) {
  __shared__ __align__(16) f16 Kt[2][32][72];     // key-major, 64 halfs + pad
  __shared__ __align__(16) f16 Vtile[2][64][40];  // d-major,  32 halfs + pad
  __shared__ __align__(16) f16 Plds[8][16][40];   // per-wave P tile

  const int tid  = threadIdx.x;
  const int lane = tid & 31;
  const int wave = tid >> 5;
  const int hi   = lane >> 4;
  const int bh   = blockIdx.x;
  const int b    = bh >> 4;
  const int h    = bh & 15;
  const int q0   = blockIdx.y * 128 + wave * 16;

  const f16* Qp = Qh + (size_t)bh * Sc * DHc;
  const f16* Kp = Kh + (size_t)bh * Sc * DHc;
  const f16* Vp = Vt + (size_t)bh * DHc * Sc;
  const int* Mp = mask + ((size_t)b * Sc + q0) * Sc;

  // cooperative loaders
  const int kRow = tid >> 3, kCol = (tid & 7) * 8;   // K: 32 rows x 64 halfs
  const int vRow = tid >> 2, vCol = (tid & 3) * 8;   // V: 64 rows x 32 halfs
  const f16* gK = Kp + (size_t)kRow * DHc + kCol;
  const f16* gV = Vp + (size_t)vRow * Sc + vCol;

  async_load_b128(&Kt[0][kRow][kCol], gK);
  async_load_b128(&Vtile[0][vRow][vCol], gV);

  // Q fragments, pre-scaled by 1/sqrt(DH)
  v16h qa0 = load_afrag(Qp, q0, 0, DHc);
  v16h qa1 = load_afrag(Qp, q0, 32, DHc);
#pragma unroll
  for (int i = 0; i < 16; ++i) {
    qa0[i] *= (f16)0.125f;
    qa1[i] *= (f16)0.125f;
  }

  float mrun[8], lrun[8];
#pragma unroll
  for (int r = 0; r < 8; ++r) { mrun[r] = -1e30f; lrun[r] = 0.0f; }
  v8f zf = {};
  v8f o[4] = {zf, zf, zf, zf};

  for (int i = 0; i < Sc / 32; ++i) {
    const int kb  = i * 32;
    const int buf = i & 1;
    wait_async0();
    __syncthreads();
    if (kb + 32 < Sc) {
      async_load_b128(&Kt[buf ^ 1][kRow][kCol], gK + (size_t)(kb + 32) * DHc);
      async_load_b128(&Vtile[buf ^ 1][vRow][vCol], gV + kb + 32);
    }

    // ---- scores S[16, kb..kb+31] from LDS K tile ----
    v8f s0 = zf, s1 = zf;
    s0 = wmma32(qa0, lds_bfrag(&Kt[buf][(lane & 15)][     hi * 16]), s0);
    s0 = wmma32(qa1, lds_bfrag(&Kt[buf][(lane & 15)][32 + hi * 16]), s0);
    s1 = wmma32(qa0, lds_bfrag(&Kt[buf][16 + (lane & 15)][     hi * 16]), s1);
    s1 = wmma32(qa1, lds_bfrag(&Kt[buf][16 + (lane & 15)][32 + hi * 16]), s1);

    // ---- mask ----
#pragma unroll
    for (int r = 0; r < 8; ++r) {
      const int mrow = r + hi * 8;
      const size_t mi = (size_t)mrow * Sc + kb + (lane & 15);
      if (Mp[mi] == 0)      s0[r] = -1e9f;
      if (Mp[mi + 16] == 0) s1[r] = -1e9f;
    }

    // ---- online softmax + stage P into LDS ----
#pragma unroll
    for (int r = 0; r < 8; ++r) {
      float mx = fmaxf(s0[r], s1[r]);
#pragma unroll
      for (int off = 1; off < 16; off <<= 1)
        mx = fmaxf(mx, __shfl_xor(mx, off, 32));
      const float nm    = fmaxf(mrun[r], mx);
      const float alpha = __expf(mrun[r] - nm);
      mrun[r] = nm;
      const float p0 = __expf(s0[r] - nm);
      const float p1 = __expf(s1[r] - nm);
      float rs = p0 + p1;
#pragma unroll
      for (int off = 1; off < 16; off <<= 1)
        rs += __shfl_xor(rs, off, 32);
      lrun[r] = lrun[r] * alpha + rs;
#pragma unroll
      for (int c = 0; c < 4; ++c) o[c][r] *= alpha;

      const int mrow = r + hi * 8;
      Plds[wave][mrow][lane & 15]        = (f16)p0;
      Plds[wave][mrow][(lane & 15) + 16] = (f16)p1;
    }

    __syncthreads();

    // ---- PV: P (A-layout from LDS) x V (B-layout from LDS V tile) ----
    {
      const f16* pp = &Plds[wave][lane & 15][hi * 8];
      v16h pa = join16(*(const v8h*)pp, *(const v8h*)(pp + 16));
#pragma unroll
      for (int c = 0; c < 4; ++c) {
        v16h vb = lds_bfrag(&Vtile[buf][c * 16 + (lane & 15)][hi * 16]);
        o[c] = wmma32(pa, vb, o[c]);
      }
    }
    __syncthreads();
  }

  // ---- normalize and store concat layout [B, S, D] f16 ----
#pragma unroll
  for (int c = 0; c < 4; ++c) {
    const int col = h * DHc + c * 16 + (lane & 15);
#pragma unroll
    for (int r = 0; r < 8; ++r) {
      const int srow = q0 + r + hi * 8;
      outc[((size_t)b * Sc + srow) * Dc + col] = (f16)(o[c][r] / lrun[r]);
    }
  }
}

// ---------------------------------------------------------------------------
// Launcher
// in order: q(0) k(1) v(2) mask(3) Wq(4) bq(5) Wk(6) bk(7) Wv(8) bv(9) Wo(10) bo(11)
// ---------------------------------------------------------------------------
extern "C" void kernel_launch(void* const* d_in, const int* in_sizes, int n_in,
                              void* d_out, int out_size, void* d_ws, size_t ws_size,
                              hipStream_t stream) {
  const float* q    = (const float*)d_in[0];
  const float* k    = (const float*)d_in[1];
  const float* v    = (const float*)d_in[2];
  const int*   mask = (const int*)d_in[3];
  const float* Wq   = (const float*)d_in[4];
  const float* bq   = (const float*)d_in[5];
  const float* Wk   = (const float*)d_in[6];
  const float* bk   = (const float*)d_in[7];
  const float* Wv   = (const float*)d_in[8];
  const float* bv   = (const float*)d_in[9];
  const float* Wo   = (const float*)d_in[10];
  const float* bo   = (const float*)d_in[11];
  float* out = (float*)d_out;

  const int NA = Mc * Dc;   // 4194304 activation elems
  const int NW = Dc * Dc;   // 1048576 weight elems

  f16* wsp  = (f16*)d_ws;
  f16* q16  = wsp;
  f16* k16  = q16 + NA;
  f16* v16  = k16 + NA;
  f16* Wq16 = v16 + NA;
  f16* Wk16 = Wq16 + NW;
  f16* Wv16 = Wk16 + NW;
  f16* Wo16 = Wv16 + NW;
  f16* Qhp  = Wo16 + NW;    // [B,H,S,DH]
  f16* Khp  = Qhp + NA;     // [B,H,S,DH]
  f16* Vtp  = Khp + NA;     // [B,H,DH,S]
  f16* cc16 = Vtp + NA;     // concat [B,S,D]

  // 1) convert to f16
  cvt_f32_f16_kernel<<<2048, 256, 0, stream>>>(q,  q16,  NA);
  cvt_f32_f16_kernel<<<2048, 256, 0, stream>>>(k,  k16,  NA);
  cvt_f32_f16_kernel<<<2048, 256, 0, stream>>>(v,  v16,  NA);
  cvt_f32_f16_kernel<<<1024, 256, 0, stream>>>(Wq, Wq16, NW);
  cvt_f32_f16_kernel<<<1024, 256, 0, stream>>>(Wk, Wk16, NW);
  cvt_f32_f16_kernel<<<1024, 256, 0, stream>>>(Wv, Wv16, NW);
  cvt_f32_f16_kernel<<<1024, 256, 0, stream>>>(Wo, Wo16, NW);

  // 2) projections
  dim3 ggrid(Mc / 256, Dc / 64);
  gemm_wmma_kernel<<<ggrid, 256, 0, stream>>>(q16, Wq16, bq, (void*)Qhp, 0);
  gemm_wmma_kernel<<<ggrid, 256, 0, stream>>>(k16, Wk16, bk, (void*)Khp, 0);
  gemm_wmma_kernel<<<ggrid, 256, 0, stream>>>(v16, Wv16, bv, (void*)Vtp, 1);

  // 3) attention
  dim3 agrid(Bc * Hc, Sc / 128);
  attn_kernel<<<agrid, 256, 0, stream>>>(Qhp, Khp, Vtp, mask, cc16);

  // 4) output projection -> fp32 d_out
  gemm_wmma_kernel<<<ggrid, 256, 0, stream>>>(cc16, Wo16, bo, (void*)out, 2);
}
